// myPCANorm_76836964925864
// MI455X (gfx1250) — matrix operand order repaired and checked by reading, not compiled, not run on hardware
//
#include <hip/hip_runtime.h>
#include <hip/hip_bf16.h>

typedef __attribute__((ext_vector_type(16))) __bf16 v16bf;
typedef __attribute__((ext_vector_type(8)))  float  v8f;

#define EPSV 1e-5f
#define C_DIM 256
#define HW 3136
#define N_IMG 32
#define M_TOT (N_IMG * HW)         /* 100352 */
#define N_EIG 32
#define N_POW 10

/* workspace layout (float offsets) */
#define WS_SUM   0
#define WS_SUMSQ 256
#define WS_MU    512
#define WS_RSIG  768
#define WS_S     1024
#define WS_V     (WS_S + 256 * 256)          /* 66560 */
#define WS_PROJ  (WS_V + N_EIG * 256)        /* 74752 */
#define WS_ZERO_N (WS_S + 256 * 256)         /* zero sums + S */

#define XROW 257  /* padded LDS row stride: 257 % 64 == 1 -> conflict-free */

// ---------------------------------------------------------------------------
__global__ void k_zero(float* ws, int n) {
  int i = blockIdx.x * blockDim.x + threadIdx.x;
  if (i < n) ws[i] = 0.f;
}

// ---------------------------------------------------------------------------
// Per-channel sum / sumsq. One block per (channel, image) pair, 3136 contig.
__global__ void __launch_bounds__(256) k_stats(const float* __restrict__ x,
                                               float* __restrict__ ws) {
  const int b = blockIdx.x;     // 0..8191
  const int c = b >> 5;
  const int n = b & 31;
  const float* p = x + ((size_t)(n * C_DIM + c)) * HW;
  float s = 0.f, ss = 0.f;
  for (int i = threadIdx.x; i < HW; i += 256) {
    float v = p[i];
    s += v; ss += v * v;
  }
  __shared__ float rs[256], rss[256];
  rs[threadIdx.x] = s; rss[threadIdx.x] = ss;
  __syncthreads();
  for (int off = 128; off > 0; off >>= 1) {
    if (threadIdx.x < off) {
      rs[threadIdx.x]  += rs[threadIdx.x + off];
      rss[threadIdx.x] += rss[threadIdx.x + off];
    }
    __syncthreads();
  }
  if (threadIdx.x == 0) {
    atomicAdd(&ws[WS_SUM + c],   rs[0]);
    atomicAdd(&ws[WS_SUMSQ + c], rss[0]);
  }
}

__global__ void k_finalize_stats(float* ws) {
  const int c = threadIdx.x;
  float s  = ws[WS_SUM + c];
  float ss = ws[WS_SUMSQ + c];
  float mu = s / (float)M_TOT;
  float var = (ss - s * mu) / (float)(M_TOT - 1);   // unbiased
  ws[WS_MU + c]   = mu;
  ws[WS_RSIG + c] = rsqrtf(var + EPSV);
}

// ---------------------------------------------------------------------------
// S += xn_chunk * xn_chunk^T  via v_wmma_f32_16x16x32_bf16, split-K over images.
// grid (16 i-tiles, 4 j-groups, 32 images), 128 threads = 4 waves, one 16x16
// output tile per wave. Normalize f32 -> bf16 on the fly.
__global__ void __launch_bounds__(128) k_cov(const float* __restrict__ x,
                                             float* __restrict__ ws) {
  const int lane = threadIdx.x & 31;
  const int wave = threadIdx.x >> 5;
  const int l15  = lane & 15;
  const int hi   = lane >> 4;                  // 0 or 1
  const int ti   = blockIdx.x;                 // row tile (0..15)
  const int tj   = blockIdx.y * 4 + wave;      // col tile (0..15)
  const int n    = blockIdx.z;                 // image

  const int ra = ti * 16 + l15;                // A channel row
  const int rb = tj * 16 + l15;                // B channel row
  const float muA = ws[WS_MU + ra], rsA = ws[WS_RSIG + ra];
  const float muB = ws[WS_MU + rb], rsB = ws[WS_RSIG + rb];
  const float2* arow = (const float2*)(x + ((size_t)(n * C_DIM + ra)) * HW);
  const float2* brow = (const float2*)(x + ((size_t)(n * C_DIM + rb)) * HW);

  v8f acc = {};
  for (int ks = 0; ks < HW / 32; ++ks) {
    const int k0 = ks * 32;
    v16bf a, b;
    // A (16x32 bf16): lanes<16 K=0..7 & 16..23; lanes>=16 K=8..15 & 24..31
    const int abase = (k0 + hi * 8) >> 1;
#pragma unroll
    for (int p = 0; p < 4; ++p) {
      float2 q  = arow[abase + p];
      a[2 * p]     = (__bf16)((q.x - muA) * rsA);
      a[2 * p + 1] = (__bf16)((q.y - muA) * rsA);
      float2 q2 = arow[abase + 8 + p];
      a[8 + 2 * p]     = (__bf16)((q2.x - muA) * rsA);
      a[8 + 2 * p + 1] = (__bf16)((q2.y - muA) * rsA);
    }
    // B (32x16 bf16): lane = N column; lanes<16 K=0..7 & 16..23; lanes>=16 K=8..15 & 24..31
    const int bbase = (k0 + hi * 8) >> 1;
#pragma unroll
    for (int p = 0; p < 8; ++p) {
      float2 q = brow[bbase + p + (p & 4)];
      b[2 * p]     = (__bf16)((q.x - muB) * rsB);
      b[2 * p + 1] = (__bf16)((q.y - muB) * rsB);
    }
    acc = __builtin_amdgcn_wmma_f32_16x16x32_bf16(false, a, false, b,
                                                  (short)0, acc, false, false);
  }
  float* S = ws + WS_S;
#pragma unroll
  for (int v = 0; v < 8; ++v) {               // C/D: VGPR v -> M = v (+8 hi)
    const int row = ti * 16 + v + hi * 8;
    const int col = tj * 16 + l15;
    atomicAdd(&S[row * C_DIM + col], acc[v]);
  }
}

// ---------------------------------------------------------------------------
// Sequential power-iteration deflation. Whole 256x256 matrix lives in LDS
// (CDNA5: 320 KB per WGP). One block, 256 threads; thread t owns row t.
__global__ void __launch_bounds__(256) k_eigen(float* ws) {
  extern __shared__ float sm[];
  float* xxt = sm;                       // XROW*256
  float* vv  = sm + XROW * 256;          // 256
  float* uu  = vv + 256;                 // 256
  float* red = uu + 256;                 // 256
  const int t = threadIdx.x;
  const float invM1 = 1.f / (float)(M_TOT - 1);
  const float* S = ws + WS_S;
  float* V = ws + WS_V;

  for (int i = 0; i < C_DIM; ++i)        // coalesced load + scale + eps*I
    xxt[i * XROW + t] = S[i * C_DIM + t] * invM1 + (i == t ? EPSV : 0.f);
  __syncthreads();

  for (int e = 0; e < N_EIG; ++e) {
    vv[t] = 1.f;
    __syncthreads();
    for (int it = 0; it < N_POW; ++it) {
      float d = 0.f;
      for (int j = 0; j < C_DIM; ++j) d += xxt[t * XROW + j] * vv[j];
      uu[t]  = d;
      red[t] = d * d;
      __syncthreads();
      for (int off = 128; off > 0; off >>= 1) {
        if (t < off) red[t] += red[t + off];
        __syncthreads();
      }
      const float nrm = fmaxf(sqrtf(red[0]), EPSV);
      vv[t] = uu[t] / nrm;
      __syncthreads();
    }
    // deflation: xxt -= (xxt @ v) v^T ; store v
    float d = 0.f;
    for (int j = 0; j < C_DIM; ++j) d += xxt[t * XROW + j] * vv[j];
    uu[t] = d;
    V[e * C_DIM + t] = vv[t];
    __syncthreads();
    for (int j = 0; j < C_DIM; ++j)
      xxt[t * XROW + j] -= uu[t] * vv[j];
    __syncthreads();
  }
}

// proj = V^T V  (tiny: 256x256 output, 32-deep reduction)
__global__ void k_proj(float* ws) {
  const int i = blockIdx.x, j = threadIdx.x;
  const float* V = ws + WS_V;
  float s = 0.f;
  for (int e = 0; e < N_EIG; ++e) s += V[e * C_DIM + i] * V[e * C_DIM + j];
  ws[WS_PROJ + i * C_DIM + j] = s;
}

// ---------------------------------------------------------------------------
// out = (proj @ xn) * weight + bias, written back in NCHW layout.
// grid (1568 m-groups, 16 c-tiles), 128 threads = 4 waves, each wave a
// 16(channel) x 16(m) tile, K = 256 (8 WMMA steps of 32).
__global__ void __launch_bounds__(128) k_out(const float* __restrict__ x,
                                             const float* __restrict__ weight,
                                             const float* __restrict__ bias,
                                             const float* __restrict__ ws,
                                             float* __restrict__ out) {
  const int lane = threadIdx.x & 31;
  const int wave = threadIdx.x >> 5;
  const int l15  = lane & 15;
  const int hi   = lane >> 4;
  const int ci   = blockIdx.y;                         // channel tile 0..15
  const long m0  = ((long)blockIdx.x * 4 + wave) * 16; // 3136 % 16 == 0
  const int  n   = (int)(m0 / HW);
  const int  hw0 = (int)(m0 - (long)n * HW);

  const float* proj = ws + WS_PROJ;
  const float* mu   = ws + WS_MU;
  const float* rsig = ws + WS_RSIG;

  const int ra = ci * 16 + l15;
  const float2* arow = (const float2*)(proj + ra * C_DIM);
  const int m = hw0 + l15;                             // this lane's column
  const float* xcol = x + ((size_t)n * C_DIM) * HW + m; // + k*HW per channel

  v8f acc = {};
#pragma unroll
  for (int kk = 0; kk < 8; ++kk) {
    const int k0 = kk * 32;
    v16bf a, b;
    const int abase = (k0 + hi * 8) >> 1;
#pragma unroll
    for (int p = 0; p < 4; ++p) {
      float2 q  = arow[abase + p];
      a[2 * p]     = (__bf16)q.x;
      a[2 * p + 1] = (__bf16)q.y;
      float2 q2 = arow[abase + 8 + p];
      a[8 + 2 * p]     = (__bf16)q2.x;
      a[8 + 2 * p + 1] = (__bf16)q2.y;
    }
    const int kb = k0 + hi * 8;
#pragma unroll
    for (int p = 0; p < 8; ++p) {
      const int k = kb + 2 * p + ((p & 4) << 1);
      float v0 = (xcol[(size_t)k * HW]       - mu[k])     * rsig[k];
      float v1 = (xcol[(size_t)(k + 1) * HW] - mu[k + 1]) * rsig[k + 1];
      b[2 * p]     = (__bf16)v0;
      b[2 * p + 1] = (__bf16)v1;
    }
    acc = __builtin_amdgcn_wmma_f32_16x16x32_bf16(false, a, false, b,
                                                  (short)0, acc, false, false);
  }
#pragma unroll
  for (int v = 0; v < 8; ++v) {
    const int c = ci * 16 + v + hi * 8;
    const float val = acc[v] * weight[c] + bias[c];
    out[((size_t)(n * C_DIM + c)) * HW + m] = val;
  }
}

// ---------------------------------------------------------------------------
extern "C" void kernel_launch(void* const* d_in, const int* in_sizes, int n_in,
                              void* d_out, int out_size, void* d_ws, size_t ws_size,
                              hipStream_t stream) {
  const float* x = (const float*)d_in[0];
  const float* w = (const float*)d_in[1];
  const float* b = (const float*)d_in[2];
  float* out = (float*)d_out;
  float* ws  = (float*)d_ws;

  k_zero<<<(WS_ZERO_N + 255) / 256, 256, 0, stream>>>(ws, WS_ZERO_N);
  k_stats<<<N_IMG * C_DIM, 256, 0, stream>>>(x, ws);
  k_finalize_stats<<<1, 256, 0, stream>>>(ws);
  k_cov<<<dim3(16, 4, 32), 128, 0, stream>>>(x, ws);
  const size_t lds = (size_t)(XROW * 256 + 3 * 256) * sizeof(float); // ~260 KB
  k_eigen<<<1, 256, lds, stream>>>(ws);
  k_proj<<<256, 256, 0, stream>>>(ws);
  k_out<<<dim3(98 * 16, 16), 128, 0, stream>>>(x, w, b, ws, out);
}
